// QLinearLR_45664092291496
// MI455X (gfx1250) — matrix operand order repaired and checked by reading, not compiled, not run on hardware
//
#include <hip/hip_runtime.h>

typedef __bf16 bf16;
typedef bf16  v16bf __attribute__((ext_vector_type(16)));
typedef bf16  v8bf  __attribute__((ext_vector_type(8)));
typedef float v8f   __attribute__((ext_vector_type(8)));
typedef float v4f   __attribute__((ext_vector_type(4)));
typedef int   v4i   __attribute__((ext_vector_type(4)));

#define IN_F   4096
#define OUT_F  4096
#define M_ROWS 8192
#define RANK   64
#define ROWH   40   // padded LDS row stride in bf16 elems (80 B -> conflict-free 16-row b128 loads)

// Keep LDS fragment loads from being interleaved (and serialized) with WMMAs.
static __device__ __forceinline__ void sched_fence() {
#if __has_builtin(__builtin_amdgcn_sched_barrier)
    __builtin_amdgcn_sched_barrier(0);
#endif
}

// QLoRA NF4 codebook (matches reference _nf_table construction, offset 0.9677083)
__device__ const float NF4_TAB[16] = {
    -1.0f,                -0.6961928009986877f, -0.5250730514526367f, -0.39491748809814453f,
    -0.28444138169288635f,-0.18477343022823334f,-0.09105003625154495f, 0.0f,
     0.07958029955625534f, 0.16093020141124725f, 0.24611230194568634f, 0.33791524171829224f,
     0.44070982933044434f, 0.5626170039176941f,  0.7229568362236023f,  1.0f };

static __device__ __forceinline__ v16bf ldfrag(const bf16* p0, const bf16* p1) {
    v8bf lo = *(const v8bf*)p0;
    v8bf hi = *(const v8bf*)p1;
    return __builtin_shufflevector(lo, hi, 0,1,2,3,4,5,6,7,8,9,10,11,12,13,14,15);
}

static __device__ __forceinline__ v8f wmma_bf16(v16bf a, v16bf b, v8f c) {
    // (neg_a, A, neg_b, B, c_mod, C, reuse_a, reuse_b)
    return __builtin_amdgcn_wmma_f32_16x16x32_bf16(false, a, false, b, (short)0, c, false, false);
}

// ---------------------------------------------------------------------------
// Kernel 1: t = x @ lora_A^T   (8192 x 64, bf16)  -- M tile 128, N = 64
// ---------------------------------------------------------------------------
__global__ __launch_bounds__(256)
void lora_t_kernel(const float* __restrict__ x, const float* __restrict__ loraA,
                   bf16* __restrict__ t_ws)
{
    __shared__ __align__(16) bf16 sX[128 * ROWH];
    __shared__ __align__(16) bf16 sA[64 * ROWH];

    const int tid  = threadIdx.x;
    const int lane = tid & 31;
    const int wid  = tid >> 5;          // 8 waves, 16 M-rows each
    const int l16  = lane & 15;
    const int half = lane >> 4;
    const int mBase = blockIdx.x * 128;

    v8f acc[4];
    #pragma unroll
    for (int j = 0; j < 4; ++j)
        #pragma unroll
        for (int r = 0; r < 8; ++r) acc[j][r] = 0.f;

    const int rowX = tid >> 1, hX = tid & 1;   // x-tile fill: 128 rows x 32 K
    const int rowA = tid >> 2, qA = tid & 3;   // A-tile fill:  64 rows x 32 K

    for (int k0 = 0; k0 < IN_F; k0 += 32) {
        __syncthreads();
        {   // x tile (f32 -> bf16)
            const v4f* xp = (const v4f*)(x + (size_t)(mBase + rowX) * IN_F + k0 + hX * 16);
            bf16 tmp[16];
            #pragma unroll
            for (int i = 0; i < 4; ++i) {
                v4f f = xp[i];
                #pragma unroll
                for (int e = 0; e < 4; ++e) tmp[4 * i + e] = (bf16)f[e];
            }
            v8bf* dst = (v8bf*)&sX[rowX * ROWH + hX * 16];
            dst[0] = *(v8bf*)&tmp[0];
            dst[1] = *(v8bf*)&tmp[8];
        }
        {   // lora_A tile (f32 -> bf16)
            const v4f* ap = (const v4f*)(loraA + (size_t)rowA * IN_F + k0 + qA * 8);
            bf16 tmp[8];
            #pragma unroll
            for (int i = 0; i < 2; ++i) {
                v4f f = ap[i];
                #pragma unroll
                for (int e = 0; e < 4; ++e) tmp[4 * i + e] = (bf16)f[e];
            }
            *(v8bf*)&sA[rowA * ROWH + qA * 8] = *(v8bf*)&tmp[0];
        }
        __syncthreads();

        // fragment-load phase: all LDS loads issue before any WMMA (single wait)
        const int m0 = wid * 16;
        v16bf af = ldfrag(&sX[(m0 + l16) * ROWH + half * 8],
                          &sX[(m0 + l16) * ROWH + 16 + half * 8]);
        v16bf bfj[4];
        #pragma unroll
        for (int j = 0; j < 4; ++j) {
            const int n0 = j * 16;
            bfj[j] = ldfrag(&sA[(n0 + l16) * ROWH + half * 8],
                            &sA[(n0 + l16) * ROWH + 16 + half * 8]);
        }
        sched_fence();
        #pragma unroll
        for (int j = 0; j < 4; ++j)
            acc[j] = wmma_bf16(af, bfj[j], acc[j]);
    }

    const int m0 = wid * 16;
    #pragma unroll
    for (int j = 0; j < 4; ++j) {
        const int n = j * 16 + l16;
        #pragma unroll
        for (int r = 0; r < 8; ++r) {
            const int m = mBase + m0 + r + half * 8;
            t_ws[(size_t)m * RANK + n] = (bf16)acc[j][r];
        }
    }
}

// ---------------------------------------------------------------------------
// Kernel 2: out = x @ dequantNF4(W)^T + t @ lora_B^T
// 128x128 tile / block; 8 waves in 4(M) x 2(N); fused in-flight NF4 dequant;
// LDS double-buffered: global loads for iter k+1 are staged in registers and
// fly under the 8 WMMAs of iter k; one barrier per K-iteration.
// LoRA folded in as two extra K=32 iterations.
// ---------------------------------------------------------------------------
__global__ __launch_bounds__(256)
void qlora_gemm_kernel(const float* __restrict__ x, const int* __restrict__ qweight,
                       const float* __restrict__ absmax, const float* __restrict__ loraB,
                       const bf16* __restrict__ t_ws, float* __restrict__ out)
{
    __shared__ __align__(16) bf16 sX[2][128 * ROWH];
    __shared__ __align__(16) bf16 sW[2][128 * ROWH];
    __shared__ float s_tab[16];

    const int tid  = threadIdx.x;
    const int lane = tid & 31;
    const int wid  = tid >> 5;
    const int l16  = lane & 15;
    const int half = lane >> 4;
    const int waveM = wid & 3;      // 4 waves x 32 rows
    const int waveN = wid >> 2;     // 2 waves x 64 cols
    const int mBase = blockIdx.y * 128;
    const int nBase = blockIdx.x * 128;

    if (tid < 16) s_tab[tid] = NF4_TAB[tid];

    v8f acc[2][4];
    #pragma unroll
    for (int i = 0; i < 2; ++i)
        #pragma unroll
        for (int j = 0; j < 4; ++j)
            #pragma unroll
            for (int r = 0; r < 8; ++r) acc[i][j][r] = 0.f;

    const int row = tid >> 1, h = tid & 1;  // tile fill: 128 rows x 32 K, 2 threads/row
    const float* xrow = x + (size_t)(mBase + row) * IN_F + h * 16;
    const size_t fb_row = (size_t)(nBase + row) * IN_F;

    // register staging for the in-flight global tile
    v4f  xs0, xs1, xs2, xs3;
    v4i  q0s, q1s;
    float ss;

    auto gload = [&](int k0) {
        const v4f* xp = (const v4f*)(xrow + k0);
        xs0 = xp[0]; xs1 = xp[1]; xs2 = xp[2]; xs3 = xp[3];
        const size_t fbase = fb_row + k0;
        const int* qp = qweight + (fbase >> 1) + h * 8;
        q0s = *(const v4i*)qp;
        q1s = *(const v4i*)(qp + 4);
        ss  = absmax[fbase >> 6];
    };

    auto lstore = [&](int p) {
        {   // x tile (f32 -> bf16)
            bf16 tmp[16];
            v4f xv[4] = { xs0, xs1, xs2, xs3 };
            #pragma unroll
            for (int i = 0; i < 4; ++i)
                #pragma unroll
                for (int e = 0; e < 4; ++e) tmp[4 * i + e] = (bf16)xv[i][e];
            v8bf* dst = (v8bf*)&sX[p][row * ROWH + h * 16];
            dst[0] = *(v8bf*)&tmp[0];
            dst[1] = *(v8bf*)&tmp[8];
        }
        {   // W tile: NF4 dequant -> bf16
            bf16 tmp[16];
            #pragma unroll
            for (int i = 0; i < 4; ++i) {
                const unsigned b = (unsigned)q0s[i];
                tmp[2 * i]     = (bf16)(s_tab[b & 15u] * ss);
                tmp[2 * i + 1] = (bf16)(s_tab[(b >> 4) & 15u] * ss);
            }
            #pragma unroll
            for (int i = 0; i < 4; ++i) {
                const unsigned b = (unsigned)q1s[i];
                tmp[8 + 2 * i]     = (bf16)(s_tab[b & 15u] * ss);
                tmp[8 + 2 * i + 1] = (bf16)(s_tab[(b >> 4) & 15u] * ss);
            }
            v8bf* dst = (v8bf*)&sW[p][row * ROWH + h * 16];
            dst[0] = *(v8bf*)&tmp[0];
            dst[1] = *(v8bf*)&tmp[8];
        }
    };

    auto compute = [&](int p) {
        const bf16* bx = &sX[p][0];
        const bf16* bw = &sW[p][0];
        v16bf af[2], bfj[4];
        #pragma unroll
        for (int i = 0; i < 2; ++i) {
            const int m0 = waveM * 32 + i * 16;
            af[i] = ldfrag(bx + (m0 + l16) * ROWH + half * 8,
                           bx + (m0 + l16) * ROWH + 16 + half * 8);
        }
        #pragma unroll
        for (int j = 0; j < 4; ++j) {
            const int n0 = waveN * 64 + j * 16;
            bfj[j] = ldfrag(bw + (n0 + l16) * ROWH + half * 8,
                            bw + (n0 + l16) * ROWH + 16 + half * 8);
        }
        sched_fence();   // all 12 ds_load_b128 above, 8 WMMAs below: one dscnt wait
        #pragma unroll
        for (int i = 0; i < 2; ++i)
            #pragma unroll
            for (int j = 0; j < 4; ++j)
                acc[i][j] = wmma_bf16(af[i], bfj[j], acc[i][j]);
    };

    __syncthreads();            // s_tab visible to all waves
    gload(0);

    // ---- main K loop, ping-pong pipelined: 1 barrier / iter ----
    int p = 0;
    for (int k0 = 0; k0 < IN_F; k0 += 32) {
        lstore(p);
        __syncthreads();
        if (k0 + 32 < IN_F) gload(k0 + 32);   // HBM latency hides under WMMAs
        compute(p);
        p ^= 1;
    }

    // ---- LoRA: two extra K=32 iterations (A <- t tile, B <- lora_B rows) ----
    for (int kk = 0; kk < RANK; kk += 32) {
        __syncthreads();
        {   // t tile (already bf16: plain copy)
            const v8bf* tp = (const v8bf*)(t_ws + (size_t)(mBase + row) * RANK + kk + h * 16);
            v8bf* dst = (v8bf*)&sX[0][row * ROWH + h * 16];
            dst[0] = tp[0];
            dst[1] = tp[1];
        }
        {   // lora_B tile (f32 -> bf16)
            const v4f* bp = (const v4f*)(loraB + (size_t)(nBase + row) * RANK + kk + h * 16);
            bf16 tmp[16];
            #pragma unroll
            for (int i = 0; i < 4; ++i) {
                v4f f = bp[i];
                #pragma unroll
                for (int e = 0; e < 4; ++e) tmp[4 * i + e] = (bf16)f[e];
            }
            v8bf* dst = (v8bf*)&sW[0][row * ROWH + h * 16];
            dst[0] = *(v8bf*)&tmp[0];
            dst[1] = *(v8bf*)&tmp[8];
        }
        __syncthreads();
        compute(0);
    }

    // ---- epilogue: f32 C tiles -> out (lane0-15 row M=r, lane16-31 row M=r+8) ----
    #pragma unroll
    for (int i = 0; i < 2; ++i) {
        const int mB = mBase + waveM * 32 + i * 16 + half * 8;
        #pragma unroll
        for (int j = 0; j < 4; ++j) {
            const int n = nBase + waveN * 64 + j * 16 + l16;
            #pragma unroll
            for (int r = 0; r < 8; ++r)
                out[(size_t)(mB + r) * OUT_F + n] = acc[i][j][r];
        }
    }
}

extern "C" void kernel_launch(void* const* d_in, const int* in_sizes, int n_in,
                              void* d_out, int out_size, void* d_ws, size_t ws_size,
                              hipStream_t stream)
{
    const float* x       = (const float*)d_in[0];
    const int*   qweight = (const int*)  d_in[1];
    const float* absmax  = (const float*)d_in[2];
    const float* loraA   = (const float*)d_in[3];
    const float* loraB   = (const float*)d_in[4];
    float*       out     = (float*)d_out;
    bf16*        t_ws    = (bf16*)d_ws;   // 8192 x 64 bf16 = 1 MB

    lora_t_kernel<<<dim3(M_ROWS / 128), 256, 0, stream>>>(x, loraA, t_ws);
    qlora_gemm_kernel<<<dim3(OUT_F / 128, M_ROWS / 128), 256, 0, stream>>>(
        x, qweight, absmax, loraB, t_ws, out);
}
